// GINEncoder_41154376630710
// MI455X (gfx1250) — matrix operand order repaired and checked by reading, not compiled, not run on hardware
//
#include <hip/hip_runtime.h>
#include <hip/hip_bf16.h>

#define N_ATOMS   100000
#define N_EDGES   800000
#define ATOM_FDIM 133
#define HIDDEN    256
#define DEPTH     3
#define NUM_MOLS  4096
#define BN_EPS    1e-5f

#define MTILE 64   // rows per workgroup (4 waves x 16-row strips)

typedef __attribute__((ext_vector_type(16))) __bf16 v16bf;
typedef __attribute__((ext_vector_type(8)))  float  v8f;

union FragA { unsigned short u[16]; v16bf v; };
union FragB { uint4 q[2];           v16bf v; };

static __device__ __forceinline__ unsigned short f2bf(float f) {
    unsigned int u = __float_as_uint(f);
    u += 0x7fffu + ((u >> 16) & 1u);          // round-to-nearest-even
    return (unsigned short)(u >> 16);
}

// ---------------------------------------------------------------------------
// Re-lay a 256x256 f32 weight matrix into per-lane bf16 WMMA B-fragment order:
//   Wf[((t*8 + s)*32 + lane)*16 + e]  ==  w[k][n],
//   k = s*32 + (lane/16)*16 + e,   n = t*16 + (lane%16)
// so each lane's 16 bf16 (32B) are contiguous -> two b128 loads at GEMM time.
// ---------------------------------------------------------------------------
__global__ __launch_bounds__(256)
void prep_wfrag(const float* __restrict__ w1, const float* __restrict__ w2,
                unsigned short* __restrict__ Wf) {
    int gid = blockIdx.x * blockDim.x + threadIdx.x;   // 6 * 65536
    int mat = gid >> 16;
    int r   = gid & 65535;
    int e = r & 15, l = (r >> 4) & 31, s = (r >> 9) & 7, t = (r >> 12) & 15;
    int k = s * 32 + (l >> 4) * 16 + e;
    int n = t * 16 + (l & 15);
    int d = mat >> 1;
    const float* w = (mat & 1) ? (w2 + d * HIDDEN * HIDDEN)
                               : (w1 + d * HIDDEN * HIDDEN);
    Wf[gid] = f2bf(w[k * HIDDEN + n]);
}

// ---------------------------------------------------------------------------
// Input layer: x = relu(f_atoms @ W_in + b).  K=133 (8% of FLOPs) -> FMA path.
// ---------------------------------------------------------------------------
__global__ __launch_bounds__(256)
void input_gemm(const float* __restrict__ F, const float* __restrict__ Ww,
                const float* __restrict__ Wb, float* __restrict__ X) {
    __shared__ float sf[ATOM_FDIM];
    int row = blockIdx.x;
    int c = threadIdx.x;
    if (c < ATOM_FDIM) sf[c] = F[(size_t)row * ATOM_FDIM + c];
    __syncthreads();
    float acc = Wb[c];
    #pragma unroll 7
    for (int k = 0; k < ATOM_FDIM; ++k)
        acc = fmaf(sf[k], Ww[k * HIDDEN + c], acc);
    X[(size_t)row * HIDDEN + c] = fmaxf(acc, 0.f);
}

// agg[dst] += x[src] : 64 threads per edge, float4 gather + 4 scalar f32 atomics
__global__ __launch_bounds__(256)
void scatter_edges(const float* __restrict__ X, float* __restrict__ Hb,
                   const int* __restrict__ ei) {
    long long gid = (long long)blockIdx.x * blockDim.x + threadIdx.x;
    if (gid >= (long long)N_EDGES * 64) return;
    int e = (int)(gid >> 6);
    int c = (int)(gid & 63) * 4;
    int src = ei[e];
    int dst = ei[N_EDGES + e];
    src = max(0, min(src, N_ATOMS - 1));
    dst = max(0, min(dst, N_ATOMS - 1));
    float4 v = *(const float4*)(X + (size_t)src * HIDDEN + c);
    float* p = Hb + (size_t)dst * HIDDEN + c;
    atomicAdd(p + 0, v.x); atomicAdd(p + 1, v.y);
    atomicAdd(p + 2, v.z); atomicAdd(p + 3, v.w);
}

// ---------------------------------------------------------------------------
// Fused GEMM: Y = f(A-inputs) @ W + b, plus per-column sum / sum-of-squares.
// The A-tile transform is applied during the f32 -> bf16 LDS staging:
//   mode 1 (GIN combine): A = P + (1 + eps[epsIdx]) * Q
//   mode 2 (BN + ReLU)  : A = relu( (P - mean) * rsqrt(var+eps)*gamma + beta )
//                          with mean/var derived from statsum/statsq (prev GEMM)
// Block = 128 threads (4 waves); each wave: 16 rows x 256 cols =
// 16 accumulator tiles, 8 K-steps of v_wmma_f32_16x16x32_bf16 each.
// ---------------------------------------------------------------------------
__global__ __launch_bounds__(128)
void gemm_fused(const float* __restrict__ P, const float* __restrict__ Q,
                const float* __restrict__ epsPtr, int epsIdx,
                const float* __restrict__ statsum, const float* __restrict__ statsq,
                const float* __restrict__ gammaP, const float* __restrict__ betaP,
                float invN, int mode,
                const unsigned short* __restrict__ Wf,
                const float* __restrict__ bias,
                float* __restrict__ Y,
                float* __restrict__ colsum, float* __restrict__ colsq) {
    __shared__ unsigned short sA[MTILE][HIDDEN + 8];
    __shared__ float sScale[HIDDEN];
    __shared__ float sShift[HIDDEN];

    const int tid  = threadIdx.x;
    const int wave = tid >> 5;
    const int lane = tid & 31;
    const int g    = lane >> 4;       // lane half
    const int idx  = lane & 15;
    const int rowb = blockIdx.x * MTILE;

    float alpha = 0.f;
    if (mode == 1) {
        alpha = 1.0f + epsPtr[epsIdx];
    } else {
        // per-column BN scale/shift from previous GEMM's accumulated stats
        for (int c = tid; c < HIDDEN; c += 128) {
            float mean = statsum[c] * invN;
            float var  = fmaxf(statsq[c] * invN - mean * mean, 0.f);
            float sc   = rsqrtf(var + BN_EPS) * gammaP[c];
            sScale[c] = sc;
            sShift[c] = betaP[c] - mean * sc;
        }
        __syncthreads();
    }

    // Stage 64x256 tile -> bf16 LDS, applying the fused transform
    #pragma unroll
    for (int i = 0; i < 32; ++i) {
        int flat = i * 512 + tid * 4;
        int r = flat >> 8, c = flat & 255;
        int gr = rowb + r;
        float4 v = make_float4(0.f, 0.f, 0.f, 0.f);
        if (gr < N_ATOMS) {
            v = *(const float4*)(P + (size_t)gr * HIDDEN + c);
            if (mode == 1) {
                float4 q = *(const float4*)(Q + (size_t)gr * HIDDEN + c);
                v.x += alpha * q.x; v.y += alpha * q.y;
                v.z += alpha * q.z; v.w += alpha * q.w;
            } else {
                v.x = fmaxf(v.x * sScale[c + 0] + sShift[c + 0], 0.f);
                v.y = fmaxf(v.y * sScale[c + 1] + sShift[c + 1], 0.f);
                v.z = fmaxf(v.z * sScale[c + 2] + sShift[c + 2], 0.f);
                v.w = fmaxf(v.w * sScale[c + 3] + sShift[c + 3], 0.f);
            }
        }
        sA[r][c + 0] = f2bf(v.x);
        sA[r][c + 1] = f2bf(v.y);
        sA[r][c + 2] = f2bf(v.z);
        sA[r][c + 3] = f2bf(v.w);
    }
    __syncthreads();

    v8f zero = {};
    v8f acc[16];
    #pragma unroll
    for (int t = 0; t < 16; ++t) acc[t] = zero;

    const int arow = wave * 16 + idx;          // A-matrix lane row (M = lane%16)
    #pragma unroll
    for (int s = 0; s < 8; ++s) {
        FragA a;
        #pragma unroll
        for (int e = 0; e < 16; ++e) {
            // A layout (matches B fragments): K = lane half * 16 + e
            int k = s * 32 + g * 16 + e;
            a.u[e] = sA[arow][k];
        }
        #pragma unroll
        for (int t = 0; t < 16; ++t) {
            FragB b;
            const uint4* bp =
                (const uint4*)(Wf + ((((t * 8 + s) * 32) + lane) << 4));
            b.q[0] = bp[0];
            b.q[1] = bp[1];
            acc[t] = __builtin_amdgcn_wmma_f32_16x16x32_bf16(
                false, a.v, false, b.v, (short)0, acc[t], false, false);
        }
    }

    // Epilogue: bias add, store, per-column partial stats (valid rows only)
    #pragma unroll
    for (int t = 0; t < 16; ++t) {
        int n = t * 16 + idx;
        float bn = bias[n];
        float s1 = 0.f, s2 = 0.f;
        #pragma unroll
        for (int r = 0; r < 8; ++r) {
            int grow = rowb + wave * 16 + r + g * 8;   // C/D: M = r + 8*half
            if (grow < N_ATOMS) {
                float y = acc[t][r] + bn;
                Y[(size_t)grow * HIDDEN + n] = y;
                s1 += y;
                s2 += y * y;
            }
        }
        atomicAdd(colsum + n, s1);
        atomicAdd(colsq  + n, s2);
    }
}

// Out = relu( (Y - mean) * rsqrt(var + eps) * gamma + beta )  (materialize x)
__global__ void bn_apply(const float* __restrict__ Y,
                         const float* __restrict__ colsum,
                         const float* __restrict__ colsq,
                         const float* __restrict__ gamma,
                         const float* __restrict__ beta,
                         float* __restrict__ Out, int n, float invN) {
    for (int i = blockIdx.x * blockDim.x + threadIdx.x; i < n;
         i += gridDim.x * blockDim.x) {
        int c = i & (HIDDEN - 1);
        float mean = colsum[c] * invN;
        float var  = fmaxf(colsq[c] * invN - mean * mean, 0.f);
        float v = (Y[i] - mean) * rsqrtf(var + BN_EPS) * gamma[c] + beta[c];
        Out[i] = fmaxf(v, 0.f);
    }
}

__global__ __launch_bounds__(256)
void seg_sum(const float* __restrict__ X, const int* __restrict__ seg,
             float* __restrict__ Out) {
    int atom = blockIdx.x;
    int c = threadIdx.x;
    int s = seg[atom];
    s = max(0, min(s, NUM_MOLS - 1));
    atomicAdd(Out + (size_t)s * HIDDEN + c, X[(size_t)atom * HIDDEN + c]);
}

__global__ void seg_count(const int* __restrict__ seg,
                          float* __restrict__ counts) {
    int a = blockIdx.x * blockDim.x + threadIdx.x;
    if (a < N_ATOMS) {
        int s = seg[a];
        s = max(0, min(s, NUM_MOLS - 1));
        atomicAdd(counts + s, 1.f);
    }
}

__global__ void seg_div(float* __restrict__ Out,
                        const float* __restrict__ counts) {
    int i = blockIdx.x * blockDim.x + threadIdx.x;   // NUM_MOLS * HIDDEN
    Out[i] /= fmaxf(counts[i >> 8], 1.f);
}

// ---------------------------------------------------------------------------
extern "C" void kernel_launch(void* const* d_in, const int* in_sizes, int n_in,
                              void* d_out, int out_size, void* d_ws,
                              size_t ws_size, hipStream_t stream) {
    const float* f_atoms = (const float*)d_in[0];
    const float* W_in_w  = (const float*)d_in[1];
    const float* W_in_b  = (const float*)d_in[2];
    const float* w1      = (const float*)d_in[3];
    const float* b1      = (const float*)d_in[4];
    const float* g1      = (const float*)d_in[5];
    const float* be1     = (const float*)d_in[6];
    const float* w2      = (const float*)d_in[7];
    const float* b2      = (const float*)d_in[8];
    const float* g2      = (const float*)d_in[9];
    const float* be2     = (const float*)d_in[10];
    const float* eps     = (const float*)d_in[11];
    const int*   ei      = (const int*)d_in[12];
    const int*   segids  = (const int*)d_in[13];
    float* out = (float*)d_out;

    // workspace layout
    const size_t SZ = (size_t)N_ATOMS * HIDDEN * sizeof(float);  // 102.4 MB
    char* ws = (char*)d_ws;
    float* bufA = (float*)(ws);               // x
    float* bufB = (float*)(ws + SZ);          // agg / Y2
    float* bufC = (float*)(ws + 2 * SZ);      // Y1
    unsigned short* Wf = (unsigned short*)(ws + 3 * SZ);  // 6*65536 bf16
    float* statsA = (float*)(ws + 3 * SZ + 6 * 65536 * sizeof(unsigned short));
    float* statsB = statsA + 2 * HIDDEN;
    float* counts = statsA + 4 * HIDDEN;

    const int nElem = N_ATOMS * HIDDEN;
    const float invN = 1.0f / (float)N_ATOMS;
    const int gemmBlocks = (N_ATOMS + MTILE - 1) / MTILE;     // 1563
    const int ewBlocks = 4096;                                 // grid-stride

    // 1) weight fragments (bf16, WMMA B layout)
    prep_wfrag<<<(6 * 65536) / 256, 256, 0, stream>>>(w1, w2, Wf);

    // 2) input layer
    input_gemm<<<N_ATOMS, HIDDEN, 0, stream>>>(f_atoms, W_in_w, W_in_b, bufA);

    // 3) message-passing layers
    for (int d = 0; d < DEPTH; ++d) {
        // agg = scatter-add of x over edges (combine with (1+eps)x fused into GEMM1)
        hipMemsetAsync(bufB, 0, SZ, stream);
        scatter_edges<<<(int)(((long long)N_EDGES * 64) / 256), 256, 0,
                        stream>>>(bufA, bufB, ei);

        // GEMM1: Y1 = (agg + (1+eps)x) @ w1 + b1 ; accumulate statsA
        hipMemsetAsync(statsA, 0, 2 * HIDDEN * sizeof(float), stream);
        gemm_fused<<<gemmBlocks, 128, 0, stream>>>(
            bufB, bufA, eps, d, nullptr, nullptr, nullptr, nullptr, invN,
            /*mode=*/1, Wf + (size_t)(d * 2 + 0) * 65536, b1 + d * HIDDEN,
            bufC, statsA, statsA + HIDDEN);

        // GEMM2: Y2 = relu(bn(Y1)) @ w2 + b2 ; accumulate statsB
        hipMemsetAsync(statsB, 0, 2 * HIDDEN * sizeof(float), stream);
        gemm_fused<<<gemmBlocks, 128, 0, stream>>>(
            bufC, nullptr, nullptr, 0, statsA, statsA + HIDDEN,
            g1 + d * HIDDEN, be1 + d * HIDDEN, invN,
            /*mode=*/2, Wf + (size_t)(d * 2 + 1) * 65536, b2 + d * HIDDEN,
            bufB, statsB, statsB + HIDDEN);

        // x = relu(bn(Y2))  (materialized: feeds gather + next combine)
        bn_apply<<<ewBlocks, 256, 0, stream>>>(bufB, statsB, statsB + HIDDEN,
                                               g2 + d * HIDDEN, be2 + d * HIDDEN,
                                               bufA, nElem, invN);
    }

    // 4) per-molecule mean
    hipMemsetAsync(out, 0, (size_t)out_size * sizeof(float), stream);
    hipMemsetAsync(counts, 0, NUM_MOLS * sizeof(float), stream);
    seg_sum<<<N_ATOMS, HIDDEN, 0, stream>>>(bufA, segids, out);
    seg_count<<<(N_ATOMS + 255) / 256, 256, 0, stream>>>(segids, counts);
    seg_div<<<(NUM_MOLS * HIDDEN) / 256, 256, 0, stream>>>(out, counts);
}
